// MLAttention_28535762715050
// MI455X (gfx1250) — matrix operand order repaired and checked by reading, not compiled, not run on hardware
//
#include <hip/hip_runtime.h>

typedef __bf16 bf16;
typedef __attribute__((ext_vector_type(16))) __bf16 v16bf;
typedef __attribute__((ext_vector_type(8)))  float  v8f;
typedef __attribute__((ext_vector_type(4)))  int    v4i_t;

constexpr int kB    = 2;
constexpr int kT    = 2048;
constexpr int kDIM  = 2048;
constexpr int kH    = 16;
constexpr int kQLR  = 1536;
constexpr int kKVLR = 512;
constexpr int kRD   = 64;
constexpr int kND   = 128;
constexpr int kVD   = 128;
constexpr int kDQK  = kND + kRD;   // 192
constexpr int kNT   = kB * kT;     // 4096 tokens total

// ---------------------------------------------------------------------------
// Optional CDNA5 async memory->LDS path (ASYNCcnt). Guarded so both the host
// pass and toolchains without the builtins fall back to synchronous staging.
// Builtin signature (from hipcc diagnostic): params are generic `int4*`
// pointers (16B payload), plus imm offset and imm cpol.
// ---------------------------------------------------------------------------
#if defined(__AMDGCN__) &&                                        \
    __has_builtin(__builtin_amdgcn_global_load_async_to_lds_b128) && \
    __has_builtin(__builtin_amdgcn_s_wait_asynccnt)
#define MLA_ASYNC_COPY 1
__device__ __forceinline__ void async_copy_b128(void* lds_dst, const void* gsrc) {
  __builtin_amdgcn_global_load_async_to_lds_b128((v4i_t*)gsrc, (v4i_t*)lds_dst,
                                                 0, 0);
}
__device__ __forceinline__ void async_wait_all() {
  __builtin_amdgcn_s_wait_asynccnt(0);
}
#else
#define MLA_ASYNC_COPY 0
#endif

// ---------------------------------------------------------------------------
// WMMA fragment loaders (CDNA5 v_wmma_f32_16x16x32_bf16 layouts, wave32)
// A (16x32, M x K): lane m = lane&15, hi = lane>>4.
//   elems 0..7  -> k = k0 + hi*8 + j        (contiguous 16B)
//   elems 8..15 -> k = k0 + 16 + hi*8 + j   (contiguous 16B)
// B (32x16, K x N) supplied as Bt[N][K] row-major: lane n = lane&15,
//   elems 0..7  -> k = k0 + hi*8 + j; elems 8..15 -> k = k0 + 16 + hi*8 + j
// C/D (16x16 f32): elem i -> m = i + 8*hi, n = lane&15.
// ---------------------------------------------------------------------------
__device__ __forceinline__ v16bf load_frag_a(const bf16* rowptr, int k0, int hi) {
  const bf16* p = rowptr + k0 + (hi << 3);
  v16bf f;
#pragma unroll
  for (int j = 0; j < 8; ++j) f[j] = p[j];
#pragma unroll
  for (int j = 0; j < 8; ++j) f[j + 8] = p[16 + j];
  return f;
}

__device__ __forceinline__ v16bf load_frag_b(const bf16* rowptr, int k0, int hi) {
  const bf16* p = rowptr + k0 + (hi << 3);
  v16bf f;
#pragma unroll
  for (int j = 0; j < 8; ++j) f[j] = p[j];
#pragma unroll
  for (int j = 0; j < 8; ++j) f[j + 8] = p[16 + j];
  return f;
}

__device__ __forceinline__ v8f wmma_bf16(v16bf a, v16bf b, v8f c) {
  return __builtin_amdgcn_wmma_f32_16x16x32_bf16(false, a, false, b, (short)0, c,
                                                 false, false);
}

// ---------------------------------------------------------------------------
// Elementwise / prep kernels
// ---------------------------------------------------------------------------
__global__ void cast_bf16_kernel(const float* __restrict__ in,
                                 bf16* __restrict__ out, long n) {
  long i = (long)blockIdx.x * blockDim.x + threadIdx.x;
  if (i < n) out[i] = (bf16)in[i];
}

// in: f32 [K][N] row-major  ->  out: bf16 [N][K] row-major
__global__ __launch_bounds__(256)
void transpose_cast_kernel(const float* __restrict__ in, bf16* __restrict__ out,
                           int K, int N) {
  __shared__ float tile[32][33];
  int kb = blockIdx.y * 32, nb = blockIdx.x * 32;
  int tx = threadIdx.x & 31, ty = threadIdx.x >> 5;  // ty 0..7
  for (int r = ty; r < 32; r += 8) {
    int k = kb + r, n = nb + tx;
    tile[r][tx] = (k < K && n < N) ? in[(size_t)k * N + n] : 0.f;
  }
  __syncthreads();
  for (int r = ty; r < 32; r += 8) {
    int n = nb + r, k = kb + tx;
    if (n < N && k < K) out[(size_t)n * K + k] = (bf16)tile[tx][r];
  }
}

// Row-wise RMSNorm: one block (256 threads) per row; writes bf16.
__global__ __launch_bounds__(256)
void rmsnorm_rows_kernel(const float* __restrict__ in, int in_stride,
                         const float* __restrict__ w, bf16* __restrict__ out,
                         int out_stride, int L) {
  int row = blockIdx.x;
  const float* x = in + (size_t)row * in_stride;
  __shared__ float red[256];
  float s = 0.f;
  for (int i = threadIdx.x; i < L; i += 256) {
    float v = x[i];
    s += v * v;
  }
  red[threadIdx.x] = s;
  __syncthreads();
  for (int off = 128; off > 0; off >>= 1) {
    if ((int)threadIdx.x < off) red[threadIdx.x] += red[threadIdx.x + off];
    __syncthreads();
  }
  float r = rsqrtf(red[0] / (float)L + 1e-6f);
  for (int i = threadIdx.x; i < L; i += 256)
    out[(size_t)row * out_stride + i] = (bf16)(x[i] * r * w[i]);
}

// Per-head RMSNorm over 128 elems: one block (128 threads) per (token, head).
__global__ __launch_bounds__(128)
void head_rmsnorm_kernel(const float* __restrict__ in, int in_tok_stride,
                         int in_head_stride, const float* __restrict__ w,
                         bf16* __restrict__ out, int out_tok_stride,
                         int out_head_stride) {
  int blk = blockIdx.x;
  int tok = blk / kH, h = blk % kH;
  const float* x = in + (size_t)tok * in_tok_stride + (size_t)h * in_head_stride;
  __shared__ float red[128];
  float v = x[threadIdx.x];
  red[threadIdx.x] = v * v;
  __syncthreads();
  for (int off = 64; off > 0; off >>= 1) {
    if ((int)threadIdx.x < off) red[threadIdx.x] += red[threadIdx.x + off];
    __syncthreads();
  }
  float r = rsqrtf(red[0] / 128.f + 1e-6f);
  out[(size_t)tok * out_tok_stride + (size_t)h * out_head_stride + threadIdx.x] =
      (bf16)(v * r * w[threadIdx.x]);
}

// RoPE for q_rope: in f32 [NT][H*RD]; out into Q bf16 [NT][H][192] at off 128.
__global__ void rope_q_kernel(const float* __restrict__ in,
                              const float* __restrict__ cosT,
                              const float* __restrict__ sinT,
                              bf16* __restrict__ Q) {
  long idx = (long)blockIdx.x * blockDim.x + threadIdx.x;
  if (idx >= (long)kNT * kH * (kRD / 2)) return;
  int i = idx & 31;
  int h = (idx >> 5) & 15;
  long tok = idx >> 9;
  int t = (int)(tok % kT);
  const float* p = in + tok * (kH * kRD) + h * kRD + 2 * i;
  float a = p[0], b = p[1];
  float c = cosT[t * 32 + i], s = sinT[t * 32 + i];
  bf16* o = Q + tok * (kH * kDQK) + h * kDQK + kND + 2 * i;
  o[0] = (bf16)(a * c - b * s);
  o[1] = (bf16)(a * s + b * c);
}

// RoPE for the shared k_rope (kv_raw cols 512..576), broadcast to all H heads.
__global__ void rope_k_kernel(const float* __restrict__ kv_raw,
                              const float* __restrict__ cosT,
                              const float* __restrict__ sinT,
                              bf16* __restrict__ K) {
  long idx = (long)blockIdx.x * blockDim.x + threadIdx.x;
  if (idx >= (long)kNT * (kRD / 2)) return;
  int i = idx & 31;
  long tok = idx >> 5;
  int t = (int)(tok % kT);
  const float* p = kv_raw + tok * (kKVLR + kRD) + kKVLR + 2 * i;
  float a = p[0], b = p[1];
  float c = cosT[t * 32 + i], s = sinT[t * 32 + i];
  bf16 e0 = (bf16)(a * c - b * s);
  bf16 e1 = (bf16)(a * s + b * c);
  bf16* base = K + tok * (kH * kDQK) + kND + 2 * i;
#pragma unroll
  for (int h = 0; h < kH; ++h) {
    base[h * kDQK] = e0;
    base[h * kDQK + 1] = e1;
  }
}

// Slice V out of kv_up result: kvup f32 [NT][H*(ND+VD)] -> V bf16 [NT][H][128].
__global__ void copy_v_kernel(const float* __restrict__ kvup,
                              bf16* __restrict__ V) {
  long idx = (long)blockIdx.x * blockDim.x + threadIdx.x;
  if (idx >= (long)kNT * kH * kVD) return;
  int d = idx & (kVD - 1);
  long th = idx >> 7;
  int h = (int)(th % kH);
  long tok = th / kH;
  V[tok * (kH * kVD) + h * kVD + d] =
      (bf16)kvup[tok * (kH * (kND + kVD)) + h * (kND + kVD) + kND + d];
}

// ---------------------------------------------------------------------------
// Generic bf16 WMMA GEMM: C[M][N] f32 = A[M][K] bf16 x Bt[N][K] bf16.
// 256 threads = 8 waves; block tile 128x128; wave tile 32x64 (2x4 WMMA tiles).
// ---------------------------------------------------------------------------
__global__ __launch_bounds__(256)
void gemm_bf16_kernel(const bf16* __restrict__ A, const bf16* __restrict__ Bt,
                      float* __restrict__ C, int M, int N, int K) {
  const int wave = threadIdx.x >> 5, lane = threadIdx.x & 31;
  const int hi = lane >> 4, nl = lane & 15;
  const int bm = blockIdx.y * 128 + (wave & 3) * 32;
  const int bn = blockIdx.x * 128 + (wave >> 2) * 64;

  const bf16* arow0 = A + (size_t)(bm + nl) * K;
  const bf16* arow1 = A + (size_t)(bm + 16 + nl) * K;
  const bf16* brow[4];
#pragma unroll
  for (int tn = 0; tn < 4; ++tn) {
    int n = bn + tn * 16 + nl;
    if (n > N - 1) n = N - 1;  // clamp for ragged N (e.g. 576)
    brow[tn] = Bt + (size_t)n * K;
  }

  v8f acc[2][4] = {};
  for (int k0 = 0; k0 < K; k0 += 32) {
    if (k0 + 128 < K) {  // L2 prefetch hint -> global_prefetch_b8
      __builtin_prefetch(arow0 + k0 + 128, 0, 3);
      __builtin_prefetch(arow1 + k0 + 128, 0, 3);
    }
    v16bf a0 = load_frag_a(arow0, k0, hi);
    v16bf a1 = load_frag_a(arow1, k0, hi);
#pragma unroll
    for (int tn = 0; tn < 4; ++tn) {
      v16bf b = load_frag_b(brow[tn], k0, hi);
      acc[0][tn] = wmma_bf16(a0, b, acc[0][tn]);
      acc[1][tn] = wmma_bf16(a1, b, acc[1][tn]);
    }
  }

#pragma unroll
  for (int tm = 0; tm < 2; ++tm)
#pragma unroll
    for (int tn = 0; tn < 4; ++tn)
#pragma unroll
      for (int i = 0; i < 8; ++i) {
        int m = bm + tm * 16 + i + 8 * hi;
        int n = bn + tn * 16 + nl;
        if (m < M && n < N) C[(size_t)m * N + n] = acc[tm][tn][i];
      }
}

// ---------------------------------------------------------------------------
// Flash attention (causal). One block per (qBlock64, head, batch).
// Q/K bf16 [NT][H][192], V bf16 [NT][H][128], O bf16 [NT][H*128].
// ---------------------------------------------------------------------------
__global__ __launch_bounds__(256)
void mla_attention_kernel(const bf16* __restrict__ Q, const bf16* __restrict__ Kc,
                          const bf16* __restrict__ V, bf16* __restrict__ O) {
  constexpr int TQ = 64, TK = 64;
  __shared__ bf16 Qs[TQ][kDQK];
  __shared__ bf16 Ks[TK][kDQK];
  __shared__ bf16 Vt[kVD][TK];   // transposed: d-major for B-fragments
  __shared__ float Sc[TQ][TK];
  __shared__ bf16 Ps[TQ][TK];
  __shared__ float m_s[TQ], l_s[TQ], c_s[TQ];
  __shared__ float pmax[TQ][4], psum[TQ][4];

  const int qb = blockIdx.x, h = blockIdx.y, bb = blockIdx.z;
  const int tid = threadIdx.x, wave = tid >> 5, lane = tid & 31;
  const int hi = lane >> 4, nl = lane & 15;
  const long tok0 = (long)bb * kT + (long)qb * TQ;
  const float scale = 0.0721687836f;  // 1/sqrt(192)

  // Stage Q (row-major, 16B chunks; kDQK=192 divisible by 8).
#if MLA_ASYNC_COPY
  for (int i = tid; i < TQ * kDQK / 8; i += 256) {
    int r = (i * 8) / kDQK, c = (i * 8) % kDQK;
    async_copy_b128(&Qs[r][c], &Q[(tok0 + r) * (kH * kDQK) + h * kDQK + c]);
  }
  async_wait_all();
#else
  for (int i = tid; i < TQ * kDQK; i += 256) {
    int r = i / kDQK, c = i % kDQK;
    Qs[r][c] = Q[(tok0 + r) * (kH * kDQK) + h * kDQK + c];
  }
#endif
  if (tid < TQ) {
    m_s[tid] = -3.0e38f;
    l_s[tid] = 0.f;
  }

  const int mt = wave & 3;   // output row tile for PV
  const int wn = wave >> 2;  // output col half for PV
  v8f o[4] = {};

  for (int kb = 0; kb <= qb; ++kb) {
    __syncthreads();  // WAR: previous iteration done with Ks/Vt/Sc/Ps
    const long ktok0 = (long)bb * kT + (long)kb * TK;
#if MLA_ASYNC_COPY
    for (int i = tid; i < TK * kDQK / 8; i += 256) {
      int r = (i * 8) / kDQK, c = (i * 8) % kDQK;
      async_copy_b128(&Ks[r][c], &Kc[(ktok0 + r) * (kH * kDQK) + h * kDQK + c]);
    }
#else
    for (int i = tid; i < TK * kDQK; i += 256) {
      int r = i / kDQK, c = i % kDQK;
      Ks[r][c] = Kc[(ktok0 + r) * (kH * kDQK) + h * kDQK + c];
    }
#endif
    for (int i = tid; i < TK * kVD; i += 256) {
      int r = i / kVD, c = i % kVD;
      Vt[c][r] = V[(ktok0 + r) * (kH * kVD) + h * kVD + c];
    }
#if MLA_ASYNC_COPY
    async_wait_all();
#endif
    __syncthreads();

    // S = Q·K^T : 16 tiles (4x4), 2 per wave, 6 k-steps over d=192.
#pragma unroll
    for (int tt = 0; tt < 2; ++tt) {
      int ti = wave * 2 + tt;
      int smt = ti >> 2, snt = ti & 3;
      const bf16* arow = &Qs[smt * 16 + nl][0];
      const bf16* brow = &Ks[snt * 16 + nl][0];
      v8f s = {};
#pragma unroll
      for (int kk = 0; kk < kDQK / 32; ++kk) {
        v16bf a = load_frag_a(arow, kk * 32, hi);
        v16bf b = load_frag_b(brow, kk * 32, hi);
        s = wmma_bf16(a, b, s);
      }
      int n = snt * 16 + nl;
      int kg = kb * TK + n;
#pragma unroll
      for (int i = 0; i < 8; ++i) {
        int m = smt * 16 + i + 8 * hi;
        int qg = qb * TQ + m;
        Sc[m][n] = (kg <= qg) ? s[i] * scale : -3.0e38f;
      }
    }
    __syncthreads();

    // Online softmax, 4 threads per row (16 cols each).
    {
      int r = tid >> 2, sub = tid & 3, c0 = sub * 16;
      float mx = -3.0e38f;
#pragma unroll
      for (int c = c0; c < c0 + 16; ++c) mx = fmaxf(mx, Sc[r][c]);
      pmax[r][sub] = mx;
    }
    __syncthreads();
    {
      int r = tid >> 2, sub = tid & 3, c0 = sub * 16;
      float gmx = fmaxf(fmaxf(pmax[r][0], pmax[r][1]),
                        fmaxf(pmax[r][2], pmax[r][3]));
      gmx = fmaxf(gmx, m_s[r]);
      float sum = 0.f;
#pragma unroll
      for (int c = c0; c < c0 + 16; ++c) {
        float p = __expf(Sc[r][c] - gmx);
        Ps[r][c] = (bf16)p;
        sum += p;
      }
      psum[r][sub] = sum;
    }
    __syncthreads();
    if (tid < TQ) {
      int r = tid;
      float gmx = fmaxf(fmaxf(pmax[r][0], pmax[r][1]),
                        fmaxf(pmax[r][2], pmax[r][3]));
      float mp = m_s[r];
      gmx = fmaxf(gmx, mp);
      float corr = __expf(mp - gmx);
      l_s[r] = l_s[r] * corr +
               (psum[r][0] + psum[r][1] + psum[r][2] + psum[r][3]);
      m_s[r] = gmx;
      c_s[r] = corr;
    }
    __syncthreads();

    // Rescale accumulators, then O += P·V (K=64 -> 2 k-steps).
#pragma unroll
    for (int j = 0; j < 4; ++j)
#pragma unroll
      for (int i = 0; i < 8; ++i) {
        int m = mt * 16 + i + 8 * hi;
        o[j][i] *= c_s[m];
      }
    const bf16* prow = &Ps[mt * 16 + nl][0];
#pragma unroll
    for (int j = 0; j < 4; ++j) {
      int nt = wn * 4 + j;
      const bf16* vrow = &Vt[nt * 16 + nl][0];
#pragma unroll
      for (int kk = 0; kk < 2; ++kk) {
        v16bf a = load_frag_a(prow, kk * 32, hi);
        v16bf b = load_frag_b(vrow, kk * 32, hi);
        o[j] = wmma_bf16(a, b, o[j]);
      }
    }
  }

  // Final normalize and store (l_s stable since last softmax+sync).
#pragma unroll
  for (int j = 0; j < 4; ++j) {
    int nt = wn * 4 + j;
#pragma unroll
    for (int i = 0; i < 8; ++i) {
      int m = mt * 16 + i + 8 * hi;
      float val = o[j][i] / l_s[m];
      O[(tok0 + m) * (kH * kVD) + h * kVD + nt * 16 + nl] = (bf16)val;
    }
  }
}

// ---------------------------------------------------------------------------
// Host-side orchestration
// ---------------------------------------------------------------------------
extern "C" void kernel_launch(void* const* d_in, const int* in_sizes, int n_in,
                              void* d_out, int out_size, void* d_ws,
                              size_t ws_size, hipStream_t stream) {
  const float* x            = (const float*)d_in[0];
  const float* fcos         = (const float*)d_in[1];
  const float* fsin         = (const float*)d_in[2];
  const float* q_down_w     = (const float*)d_in[3];
  const float* q_norm_w     = (const float*)d_in[4];
  const float* q_up_nope_w  = (const float*)d_in[5];
  const float* q_up_rope_w  = (const float*)d_in[6];
  const float* kv_down_w    = (const float*)d_in[7];
  const float* kv_norm_w    = (const float*)d_in[8];
  const float* kv_up_w      = (const float*)d_in[9];
  const float* q_head_norm_w= (const float*)d_in[10];
  const float* k_head_norm_w= (const float*)d_in[11];
  const float* wo_w         = (const float*)d_in[12];
  float* out = (float*)d_out;

  char* ws = (char*)d_ws;
  size_t off = 0;
  auto carve = [&](size_t bytes) -> void* {
    void* p = ws + off;
    off += (bytes + 255) & ~(size_t)255;
    return p;
  };

  bf16* xb       = (bf16*)carve((size_t)kNT * kDIM * 2);
  bf16* qdw_t    = (bf16*)carve((size_t)kQLR * kDIM * 2);
  bf16* qun_t    = (bf16*)carve((size_t)(kH * kND) * kQLR * 2);
  bf16* qur_t    = (bf16*)carve((size_t)(kH * kRD) * kQLR * 2);
  bf16* kvd_t    = (bf16*)carve((size_t)(kKVLR + kRD) * kDIM * 2);
  bf16* kvu_t    = (bf16*)carve((size_t)(kH * (kND + kVD)) * kKVLR * 2);
  bf16* wo_t     = (bf16*)carve((size_t)kDIM * (kH * kVD) * 2);
  float* cq_raw  = (float*)carve((size_t)kNT * kQLR * 4);
  bf16* cq       = (bf16*)carve((size_t)kNT * kQLR * 2);
  float* qn_raw  = (float*)carve((size_t)kNT * (kH * kND) * 4);
  bf16* Qb       = (bf16*)carve((size_t)kNT * kH * kDQK * 2);
  float* qr_raw  = (float*)carve((size_t)kNT * (kH * kRD) * 4);
  float* kv_raw  = (float*)carve((size_t)kNT * (kKVLR + kRD) * 4);
  bf16* ckv      = (bf16*)carve((size_t)kNT * kKVLR * 2);
  float* kvu_raw = (float*)carve((size_t)kNT * (kH * (kND + kVD)) * 4);
  bf16* Kb       = (bf16*)carve((size_t)kNT * kH * kDQK * 2);
  bf16* Vb       = (bf16*)carve((size_t)kNT * kH * kVD * 2);
  bf16* Ob       = (bf16*)carve((size_t)kNT * kH * kVD * 2);

  // 1) cast x, transpose-cast weights to bf16 [N][K]
  {
    long n = (long)kNT * kDIM;
    cast_bf16_kernel<<<(n + 255) / 256, 256, 0, stream>>>(x, xb, n);
  }
  auto tc = [&](const float* w, bf16* wt, int K, int N) {
    dim3 g((N + 31) / 32, (K + 31) / 32);
    transpose_cast_kernel<<<g, 256, 0, stream>>>(w, wt, K, N);
  };
  tc(q_down_w,    qdw_t, kDIM,  kQLR);
  tc(q_up_nope_w, qun_t, kQLR,  kH * kND);
  tc(q_up_rope_w, qur_t, kQLR,  kH * kRD);
  tc(kv_down_w,   kvd_t, kDIM,  kKVLR + kRD);
  tc(kv_up_w,     kvu_t, kKVLR, kH * (kND + kVD));
  tc(wo_w,        wo_t,  kH * kVD, kDIM);

  auto gemm = [&](const bf16* A, const bf16* Bt, float* C, int M, int N, int K) {
    dim3 g((N + 127) / 128, (M + 127) / 128);
    gemm_bf16_kernel<<<g, 256, 0, stream>>>(A, Bt, C, M, N, K);
  };

  // 2) c_q = rmsnorm(x @ q_down_w)
  gemm(xb, qdw_t, cq_raw, kNT, kQLR, kDIM);
  rmsnorm_rows_kernel<<<kNT, 256, 0, stream>>>(cq_raw, kQLR, q_norm_w, cq, kQLR,
                                               kQLR);

  // 3) q_nope = per-head rmsnorm(c_q @ q_up_nope_w) -> Q[..., 0:128]
  gemm(cq, qun_t, qn_raw, kNT, kH * kND, kQLR);
  head_rmsnorm_kernel<<<kNT * kH, 128, 0, stream>>>(
      qn_raw, kH * kND, kND, q_head_norm_w, Qb, kH * kDQK, kDQK);

  // 4) q_rope = rope(c_q @ q_up_rope_w) -> Q[..., 128:192]
  gemm(cq, qur_t, qr_raw, kNT, kH * kRD, kQLR);
  {
    long n = (long)kNT * kH * (kRD / 2);
    rope_q_kernel<<<(n + 255) / 256, 256, 0, stream>>>(qr_raw, fcos, fsin, Qb);
  }

  // 5) kv_raw = x @ kv_down_w ; c_kv = rmsnorm(kv_raw[:, :512]) ;
  //    k_rope = rope(kv_raw[:, 512:576]) broadcast -> K[..., 128:192]
  gemm(xb, kvd_t, kv_raw, kNT, kKVLR + kRD, kDIM);
  rmsnorm_rows_kernel<<<kNT, 256, 0, stream>>>(kv_raw, kKVLR + kRD, kv_norm_w,
                                               ckv, kKVLR, kKVLR);
  {
    long n = (long)kNT * (kRD / 2);
    rope_k_kernel<<<(n + 255) / 256, 256, 0, stream>>>(kv_raw, fcos, fsin, Kb);
  }

  // 6) kv = c_kv @ kv_up_w ; k_nope = per-head rmsnorm -> K[..., 0:128] ; V
  gemm(ckv, kvu_t, kvu_raw, kNT, kH * (kND + kVD), kKVLR);
  head_rmsnorm_kernel<<<kNT * kH, 128, 0, stream>>>(
      kvu_raw, kH * (kND + kVD), kND + kVD, k_head_norm_w, Kb, kH * kDQK, kDQK);
  {
    long n = (long)kNT * kH * kVD;
    copy_v_kernel<<<(n + 255) / 256, 256, 0, stream>>>(kvu_raw, Vb);
  }

  // 7) causal flash attention -> Ob bf16 [NT][H*128]
  {
    dim3 g(kT / 64, kH, kB);
    mla_attention_kernel<<<g, 256, 0, stream>>>(Qb, Kb, Vb, Ob);
  }

  // 8) out = Ob @ wo_w  (f32 result straight into d_out)
  gemm(Ob, wo_t, out, kNT, kDIM, kH * kVD);
}
